// pointnet2_62388694942115
// MI455X (gfx1250) — compile-verified, matrix operand order checked by reading, not run on hardware
//
#include <hip/hip_runtime.h>

// ---------------------------------------------------------------------------
// CDNA5 (gfx1250) PointNet++ part-seg forward.
// GEMMs use v_wmma_f32_16x16x32_bf16 with 2x2 register blocking per wave
// (32x32 output / wave, 64x64 / block) -> 2 b128 loads per WMMA.
// Everything else is bandwidth/latency-bound streaming kernels.
// ---------------------------------------------------------------------------

typedef __attribute__((ext_vector_type(16))) __bf16 v16bf;
typedef __attribute__((ext_vector_type(8)))  __bf16 v8bf;
typedef __attribute__((ext_vector_type(8)))  float  v8f;

static __device__ __forceinline__ __bf16 f2bf(float f) {
  unsigned int u = __builtin_bit_cast(unsigned int, f);
  unsigned int r = (u + 0x7FFFu + ((u >> 16) & 1u)) >> 16;
  unsigned short s = (unsigned short)r;
  return __builtin_bit_cast(__bf16, s);
}
static __device__ __forceinline__ __bf16 bf0() {
  return __builtin_bit_cast(__bf16, (unsigned short)0);
}

// Load one 16-element bf16 fragment: two contiguous 16B groups 32B apart
// (K = base+0..7 and K = base+16..23), per ISA 16-bit A/B layout.
static __device__ __forceinline__ v16bf ldfrag(const __bf16* __restrict__ p) {
  const v8bf* q = (const v8bf*)p;
  v8bf lo = q[0];
  v8bf hi = q[2];
  v16bf f;
#pragma unroll
  for (int i = 0; i < 8; ++i) { f[i] = lo[i]; f[i + 8] = hi[i]; }
  return f;
}

// ------------------------- layout / transpose kernels ----------------------

// in: (B,9,N) -> l0p (B,N,9), l0x (B,N,3)
__global__ void k_transpose_in(const float* __restrict__ in, float* __restrict__ l0p,
                               float* __restrict__ l0x, int B, int N) {
  long i = (long)blockIdx.x * blockDim.x + threadIdx.x;
  if (i >= (long)B * N) return;
  int n = (int)(i % N);
  int b = (int)(i / N);
  const float* src = in + (long)b * 9 * N;
  for (int k = 0; k < 9; ++k) l0p[i * 9 + k] = src[(long)k * N + n];
  for (int k = 0; k < 3; ++k) l0x[i * 3 + k] = src[(long)k * N + n];
}

// Y (B,N,128) -> out (B,128,N)
__global__ void k_transpose_out(const float* __restrict__ Y, float* __restrict__ out,
                                int B, int N) {
  long i = (long)blockIdx.x * blockDim.x + threadIdx.x;
  long tot = (long)B * 128 * N;
  if (i >= tot) return;
  int n = (int)(i % N);
  long t = i / N;
  int c = (int)(t % 128);
  int b = (int)(t / 128);
  out[i] = Y[((long)b * N + n) * 128 + c];
}

__global__ void k_copyf(const float* __restrict__ src, float* __restrict__ dst, long n) {
  long i = (long)blockIdx.x * blockDim.x + threadIdx.x;
  if (i < n) dst[i] = src[i];
}

// ------------------------------- FPS ---------------------------------------

#define FPS_T 512
__global__ void k_fps(const float* __restrict__ xyz, int* __restrict__ fidx,
                      int N, int npoint) {
  int b = blockIdx.x;
  int t = threadIdx.x;
  const float* xb = xyz + (long)b * N * 3;
  __shared__ float sv[FPS_T];
  __shared__ int   si[FPS_T];
  const int P = N / FPS_T;   // 8 for N=4096, 1 for N=512
  float dist[8];
  for (int p = 0; p < P; ++p) dist[p] = 1e10f;
  int far = 0;
  for (int it = 0; it < npoint; ++it) {
    if (t == 0) fidx[(long)b * npoint + it] = far;
    float cx = xb[(long)far * 3 + 0];
    float cy = xb[(long)far * 3 + 1];
    float cz = xb[(long)far * 3 + 2];
    float best = -1.0f; int bi = 0;
    for (int p = 0; p < P; ++p) {
      int j = t + p * FPS_T;
      float dx = xb[(long)j * 3 + 0] - cx;
      float dy = xb[(long)j * 3 + 1] - cy;
      float dz = xb[(long)j * 3 + 2] - cz;
      float d = dx * dx + dy * dy + dz * dz;
      dist[p] = fminf(dist[p], d);
      if (dist[p] > best) { best = dist[p]; bi = j; }
    }
    sv[t] = best; si[t] = bi;
    __syncthreads();
    for (int s = FPS_T / 2; s > 0; s >>= 1) {
      if (t < s) {
        if (sv[t + s] > sv[t] || (sv[t + s] == sv[t] && si[t + s] < si[t])) {
          sv[t] = sv[t + s]; si[t] = si[t + s];
        }
      }
      __syncthreads();
    }
    far = si[0];
    __syncthreads();
  }
}

__global__ void k_gather_xyz(const float* __restrict__ xyz, const int* __restrict__ fidx,
                             float* __restrict__ nxyz, int Nsrc, int np, int B) {
  long i = (long)blockIdx.x * blockDim.x + threadIdx.x;
  if (i >= (long)B * np) return;
  int b = (int)(i / np);
  int f = fidx[i];
  for (int k = 0; k < 3; ++k)
    nxyz[i * 3 + k] = xyz[((long)b * Nsrc + f) * 3 + k];
}

// ------------------------------ ball query ---------------------------------

__global__ void k_ballquery(const float* __restrict__ xyz, const float* __restrict__ nxyz,
                            int* __restrict__ gidx, int N, int np, int ns, float r2) {
  int b = blockIdx.y;
  int m = blockIdx.x * blockDim.x + threadIdx.x;
  if (m >= np) return;
  const float* xb = xyz + (long)b * N * 3;
  long mi = (long)b * np + m;
  float cx = nxyz[mi * 3 + 0], cy = nxyz[mi * 3 + 1], cz = nxyz[mi * 3 + 2];
  int* g = gidx + mi * ns;
  int cnt = 0;
  for (int j = 0; j < N && cnt < ns; ++j) {
    float dx = xb[(long)j * 3 + 0] - cx;
    float dy = xb[(long)j * 3 + 1] - cy;
    float dz = xb[(long)j * 3 + 2] - cz;
    float d = dx * dx + dy * dy + dz * dz;
    if (d <= r2) g[cnt++] = j;
  }
  if (cnt == 0) {
    for (int s = 0; s < ns; ++s) g[s] = N - 1;
  } else {
    for (int s = cnt; s < ns; ++s) g[s] = g[0];
  }
}

// --------------------------- grouping (-> bf16 X) --------------------------

// rows: ((b*np+m)*ns+s), cols: [0,3)=rel xyz, [3,3+C)=pts, rest zero pad
__global__ void k_group_sa(const float* __restrict__ xyz, const float* __restrict__ nxyz,
                           const int* __restrict__ gidx, const float* __restrict__ pts,
                           __bf16* __restrict__ X, int Nsrc, int np, int ns, int C,
                           int kp, int B) {
  long i = (long)blockIdx.x * blockDim.x + threadIdx.x;
  long tot = (long)B * np * ns * kp;
  if (i >= tot) return;
  int col = (int)(i % kp);
  long t = i / kp;
  int s = (int)(t % ns); t /= ns;
  int m = (int)(t % np);
  int b = (int)(t / np);
  long mi = (long)b * np + m;
  int g = gidx[mi * ns + s];
  float v;
  if (col < 3) {
    v = xyz[((long)b * Nsrc + g) * 3 + col] - nxyz[mi * 3 + col];
  } else if (col < 3 + C) {
    v = pts[((long)b * Nsrc + g) * C + (col - 3)];
  } else {
    v = 0.0f;
  }
  X[i] = f2bf(v);
}

// sa3 group_all: rows = b*128+j, cols [0,3)=l2_xyz, [3,259)=l2_points, pad
__global__ void k_group_all(const float* __restrict__ xyz, const float* __restrict__ pts,
                            __bf16* __restrict__ X, int Npts, int C, int kp, int B) {
  long i = (long)blockIdx.x * blockDim.x + threadIdx.x;
  long tot = (long)B * Npts * kp;
  if (i >= tot) return;
  int col = (int)(i % kp);
  long row = i / kp;   // b*Npts + j
  float v;
  if (col < 3)           v = xyz[row * 3 + col];
  else if (col < 3 + C)  v = pts[row * C + (col - 3)];
  else                   v = 0.0f;
  X[i] = f2bf(v);
}

// ------------------------------ WMMA GEMM ----------------------------------
// Y[M,Cout] = X[M,K](bf16) * W[Cout,K](bf16)^T + bias.
// K % 32 == 0, M % 64 == 0, Cout % 64 == 0.
// 4 waves/block arranged 2x2; each wave owns a 32x32 output macro-tile
// (2x2 WMMA tiles, 4 accumulators). 8 b128 loads feed 4 WMMAs per K-step.

__global__ void __launch_bounds__(128)
k_gemm_bf16_wmma(const __bf16* __restrict__ X, const __bf16* __restrict__ W,
                 const float* __restrict__ bias, float* __restrict__ Y,
                 long M, int K, int Cout) {
  int lane = threadIdx.x & 31;
  int wave = threadIdx.x >> 5;
  int r = lane & 15;
  int h = lane >> 4;
  int wm = wave & 1;
  int wn = wave >> 1;
  long m0 = (long)blockIdx.x * 64 + wm * 32;
  int  n0 = blockIdx.y * 64 + wn * 32;

  v8f acc00 = {}, acc01 = {}, acc10 = {}, acc11 = {};
  const __bf16* xr0 = X + (m0 + r) * (long)K;
  const __bf16* xr1 = X + (m0 + 16 + r) * (long)K;
  const __bf16* wr0 = W + ((long)(n0 + r)) * (long)K;
  const __bf16* wr1 = W + ((long)(n0 + 16 + r)) * (long)K;

  for (int k0 = 0; k0 < K; k0 += 32) {
    int ko = k0 + 8 * h;
    v16bf a0 = ldfrag(xr0 + ko);
    v16bf a1 = ldfrag(xr1 + ko);
    v16bf b0 = ldfrag(wr0 + ko);
    v16bf b1 = ldfrag(wr1 + ko);
    acc00 = __builtin_amdgcn_wmma_f32_16x16x32_bf16(false, a0, false, b0, (short)0, acc00, false, false);
    acc01 = __builtin_amdgcn_wmma_f32_16x16x32_bf16(false, a0, false, b1, (short)0, acc01, false, false);
    acc10 = __builtin_amdgcn_wmma_f32_16x16x32_bf16(false, a1, false, b0, (short)0, acc10, false, false);
    acc11 = __builtin_amdgcn_wmma_f32_16x16x32_bf16(false, a1, false, b1, (short)0, acc11, false, false);
  }

  int c0 = n0 + r;
  int c1 = n0 + 16 + r;
  float bv0 = bias[c0];
  float bv1 = bias[c1];
#pragma unroll
  for (int v = 0; v < 8; ++v) {
    long row0 = m0 + v + 8 * h;
    long row1 = row0 + 16;
    Y[row0 * (long)Cout + c0] = acc00[v] + bv0;
    Y[row0 * (long)Cout + c1] = acc01[v] + bv1;
    Y[row1 * (long)Cout + c0] = acc10[v] + bv0;
    Y[row1 * (long)Cout + c1] = acc11[v] + bv1;
  }
}

// ------------------------------ batch norm ---------------------------------

__global__ void k_bn_stats(const float* __restrict__ Y, float* __restrict__ mean,
                           float* __restrict__ var, long M, int C) {
  int c = blockIdx.x;
  __shared__ float s1[256], s2[256];
  float sum = 0.0f, sq = 0.0f;
  for (long i = threadIdx.x; i < M; i += 256) {
    float x = Y[i * C + c];
    sum += x; sq += x * x;
  }
  s1[threadIdx.x] = sum; s2[threadIdx.x] = sq;
  __syncthreads();
  for (int s = 128; s > 0; s >>= 1) {
    if ((int)threadIdx.x < s) {
      s1[threadIdx.x] += s1[threadIdx.x + s];
      s2[threadIdx.x] += s2[threadIdx.x + s];
    }
    __syncthreads();
  }
  if (threadIdx.x == 0) {
    float m = s1[0] / (float)M;
    mean[c] = m;
    var[c]  = s2[0] / (float)M - m * m;
  }
}

__global__ void k_bn_apply_relu(float* __restrict__ Y, const float* __restrict__ mean,
                                const float* __restrict__ var, const float* __restrict__ gamma,
                                const float* __restrict__ beta, long M, int C) {
  long i = (long)blockIdx.x * blockDim.x + threadIdx.x;
  if (i >= M * (long)C) return;
  int c = (int)(i % C);
  float x = Y[i];
  float y = (x - mean[c]) * rsqrtf(var[c] + 1e-5f) * gamma[c] + beta[c];
  Y[i] = fmaxf(y, 0.0f);
}

// float [M,C] -> bf16 [M,kp] zero-padded
__global__ void k_f2bf_pad(const float* __restrict__ Y, __bf16* __restrict__ X,
                           long M, int C, int kp) {
  long i = (long)blockIdx.x * blockDim.x + threadIdx.x;
  if (i >= M * (long)kp) return;
  int k = (int)(i % kp);
  long row = i / kp;
  X[i] = (k < C) ? f2bf(Y[row * C + k]) : bf0();
}

__global__ void k_wconvert(const float* __restrict__ W, __bf16* __restrict__ Wb,
                           int cout, int cin, int kp) {
  long i = (long)blockIdx.x * blockDim.x + threadIdx.x;
  long n = (long)cout * kp;
  if (i >= n) return;
  int k = (int)(i % kp);
  int o = (int)(i / kp);
  Wb[i] = (k < cin) ? f2bf(W[(long)o * cin + k]) : bf0();
}

// ------------------------------ max pooling --------------------------------

__global__ void k_maxpool(const float* __restrict__ Y, float* __restrict__ out,
                          long G, int ns, int C) {
  long i = (long)blockIdx.x * blockDim.x + threadIdx.x;
  if (i >= G * (long)C) return;
  int c = (int)(i % C);
  long g = i / C;
  float mx = -3.4e38f;
  for (int s = 0; s < ns; ++s)
    mx = fmaxf(mx, Y[((long)g * ns + s) * (long)C + c]);
  out[i] = mx;
}

// ----------------------- feature-prop input builder ------------------------
// For each (b,n): 3-NN interp from points2 (or broadcast if S==1), then
// write bf16 X row = [points1 | interp | zero-pad].

__global__ void k_fp_build(const float* __restrict__ xyz1, const float* __restrict__ xyz2,
                           const float* __restrict__ pts1, const float* __restrict__ pts2,
                           __bf16* __restrict__ X, int B, int N1, int S, int C1, int C2,
                           int kp) {
  long i = (long)blockIdx.x * blockDim.x + threadIdx.x;
  if (i >= (long)B * N1) return;
  int b = (int)(i / N1);
  float w0 = 1.0f, w1 = 0.0f, w2 = 0.0f;
  int i0 = 0, i1 = 0, i2 = 0;
  if (S > 1) {
    float px = xyz1[i * 3 + 0], py = xyz1[i * 3 + 1], pz = xyz1[i * 3 + 2];
    float d0 = 3.4e38f, d1 = 3.4e38f, d2 = 3.4e38f;
    const float* x2 = xyz2 + (long)b * S * 3;
    for (int j = 0; j < S; ++j) {
      float dx = px - x2[(long)j * 3 + 0];
      float dy = py - x2[(long)j * 3 + 1];
      float dz = pz - x2[(long)j * 3 + 2];
      float d = dx * dx + dy * dy + dz * dz;
      if (d < d0)      { d2 = d1; i2 = i1; d1 = d0; i1 = i0; d0 = d; i0 = j; }
      else if (d < d1) { d2 = d1; i2 = i1; d1 = d; i1 = j; }
      else if (d < d2) { d2 = d; i2 = j; }
    }
    float r0 = 1.0f / (d0 + 1e-8f);
    float r1 = 1.0f / (d1 + 1e-8f);
    float r2 = 1.0f / (d2 + 1e-8f);
    float rs = r0 + r1 + r2;
    w0 = r0 / rs; w1 = r1 / rs; w2 = r2 / rs;
  }
  const float* p2 = pts2 + (long)b * S * C2;
  __bf16* xr = X + i * (long)kp;
  const float* p1 = pts1 + i * (long)C1;
  for (int c = 0; c < C1; ++c) xr[c] = f2bf(p1[c]);
  for (int c = 0; c < C2; ++c) {
    float v = w0 * p2[(long)i0 * C2 + c] + w1 * p2[(long)i1 * C2 + c] +
              w2 * p2[(long)i2 * C2 + c];
    xr[C1 + c] = f2bf(v);
  }
  for (int c = C1 + C2; c < kp; ++c) xr[c] = bf0();
}

// p1 = concat([cls(14), l0_xyz(3), l0_points(9)]) -> (B,N,26)
__global__ void k_p1_build(const float* __restrict__ cls, const float* __restrict__ l0x,
                           const float* __restrict__ l0p, float* __restrict__ p1,
                           int B, int N) {
  long i = (long)blockIdx.x * blockDim.x + threadIdx.x;
  long tot = (long)B * N * 26;
  if (i >= tot) return;
  int c = (int)(i % 26);
  long t = i / 26;   // b*N+n
  int n = (int)(t % N);
  int b = (int)(t / N);
  float v;
  if (c < 14)      v = cls[((long)b * 14 + c) * N + n];
  else if (c < 17) v = l0x[t * 3 + (c - 14)];
  else             v = l0p[t * 9 + (c - 17)];
  p1[i] = v;
}

// ---------------------------------------------------------------------------

static inline unsigned gs(long n, int blk) { return (unsigned)((n + blk - 1) / blk); }
static inline int ceil32(int x) { return (x + 31) & ~31; }

extern "C" void kernel_launch(void* const* d_in, const int* in_sizes, int n_in,
                              void* d_out, int out_size, void* d_ws, size_t ws_size,
                              hipStream_t stream) {
  const int N = 4096;
  const int B = in_sizes[0] / (9 * N);

  const float* xyz_in = (const float*)d_in[0];
  const float* cls_in = (const float*)d_in[1];

  // params flatten order (sorted dict keys): fp1(3), fp2(2), fp3(2),
  // sa1(3), sa2(3), sa3(3); per layer: W, b, beta, gamma
  const float *LW[16], *LB[16], *LBETA[16], *LGAMMA[16];
  for (int li = 0; li < 16; ++li) {
    int base = 2 + li * 4;
    LW[li]     = (const float*)d_in[base + 0];
    LB[li]     = (const float*)d_in[base + 1];
    LBETA[li]  = (const float*)d_in[base + 2];
    LGAMMA[li] = (const float*)d_in[base + 3];
  }
  static const int LCOUT[16] = {128,128,128, 256,128, 256,256, 64,64,128, 128,128,256, 256,512,1024};
  static const int LCIN[16]  = {154,128,128, 384,256, 1280,256, 12,64,64, 131,128,128, 259,256,256};

  // ---------------- workspace bump allocator ----------------
  size_t off = 0;
  char* base = (char*)d_ws;
  auto alloc = [&](size_t bytes) -> void* {
    off = (off + 255) & ~(size_t)255;
    void* p = base + off;
    off += bytes;
    return p;
  };

  float* l0p    = (float*)alloc((size_t)B * N * 9 * 4);
  float* l0x    = (float*)alloc((size_t)B * N * 3 * 4);
  int*   fidx1  = (int*)  alloc((size_t)B * 512 * 4);
  float* nx1    = (float*)alloc((size_t)B * 512 * 3 * 4);      // l1_xyz
  int*   gidx1  = (int*)  alloc((size_t)B * 512 * 32 * 4);
  float* l1pts  = (float*)alloc((size_t)B * 512 * 128 * 4);
  int*   fidx2  = (int*)  alloc((size_t)B * 128 * 4);
  float* nx2    = (float*)alloc((size_t)B * 128 * 3 * 4);      // l2_xyz
  int*   gidx2  = (int*)  alloc((size_t)B * 128 * 64 * 4);
  float* l2pts  = (float*)alloc((size_t)B * 128 * 256 * 4);
  float* l3pts  = (float*)alloc((size_t)B * 1024 * 4);
  float* fp3out = (float*)alloc((size_t)B * 128 * 256 * 4);
  float* fp2out = (float*)alloc((size_t)B * 512 * 128 * 4);
  float* p1buf  = (float*)alloc((size_t)B * N * 26 * 4);
  float* meanb  = (float*)alloc(1024 * 4);
  float* varb   = (float*)alloc(1024 * 4);
  __bf16* Wbf   = (__bf16*)alloc((size_t)1024 * 1280 * 2);
  // X: max M*Kpad = (B*128*64) * 160 ; Y: max M*Cout = (B*512*32) * 128
  size_t maxX = (size_t)B * 128 * 64 * 160;
  size_t maxY = (size_t)B * 512 * 32 * 128;
  __bf16* Xbuf = (__bf16*)alloc(maxX * 2);
  float*  Ybuf = (float*) alloc(maxY * 4);
  (void)ws_size; (void)n_in; (void)out_size;

  // ---------------- MLP runner (bf16 WMMA GEMM + BN + ReLU) ----------------
  auto run_mlp = [&](long M, int li0, int nl) {
    const __bf16* Xcur = Xbuf;
    for (int l = 0; l < nl; ++l) {
      int li = li0 + l;
      int cout = LCOUT[li], cin = LCIN[li];
      int kp = ceil32(cin);
      long wn = (long)cout * kp;
      k_wconvert<<<gs(wn, 256), 256, 0, stream>>>(LW[li], Wbf, cout, cin, kp);
      dim3 g((unsigned)(M / 64), (unsigned)(cout / 64));
      k_gemm_bf16_wmma<<<g, 128, 0, stream>>>(Xcur, Wbf, LB[li], Ybuf, M, kp, cout);
      k_bn_stats<<<cout, 256, 0, stream>>>(Ybuf, meanb, varb, M, cout);
      long tot = M * cout;
      k_bn_apply_relu<<<gs(tot, 256), 256, 0, stream>>>(Ybuf, meanb, varb,
                                                        LGAMMA[li], LBETA[li], M, cout);
      if (l < nl - 1) {
        int kp2 = ceil32(cout);
        k_f2bf_pad<<<gs(M * kp2, 256), 256, 0, stream>>>(Ybuf, Xbuf, M, cout, kp2);
        Xcur = Xbuf;
      }
    }
  };

  // ---------------- stage 0: transpose input ----------------
  k_transpose_in<<<gs((long)B * N, 256), 256, 0, stream>>>(xyz_in, l0p, l0x, B, N);

  // ---------------- SA1: npoint=512, nsample=32, r=0.2, 12->[64,64,128] ----
  k_fps<<<B, FPS_T, 0, stream>>>(l0x, fidx1, N, 512);
  k_gather_xyz<<<gs((long)B * 512, 256), 256, 0, stream>>>(l0x, fidx1, nx1, N, 512, B);
  {
    dim3 g(gs(512, 128), B);
    k_ballquery<<<g, 128, 0, stream>>>(l0x, nx1, gidx1, N, 512, 32, 0.04f);
  }
  {
    long tot = (long)B * 512 * 32 * 32;  // kp=32 for cin=12
    k_group_sa<<<gs(tot, 256), 256, 0, stream>>>(l0x, nx1, gidx1, l0p, Xbuf,
                                                 N, 512, 32, 9, 32, B);
  }
  long M1 = (long)B * 512 * 32;
  run_mlp(M1, 7, 3);                       // sa1 -> Ybuf [M1,128]
  k_maxpool<<<gs((long)B * 512 * 128, 256), 256, 0, stream>>>(Ybuf, l1pts,
                                                              (long)B * 512, 32, 128);

  // ---------------- SA2: npoint=128, nsample=64, r=0.4, 131->[128,128,256] -
  k_fps<<<B, FPS_T, 0, stream>>>(nx1, fidx2, 512, 128);
  k_gather_xyz<<<gs((long)B * 128, 256), 256, 0, stream>>>(nx1, fidx2, nx2, 512, 128, B);
  {
    dim3 g(gs(128, 128), B);
    k_ballquery<<<g, 128, 0, stream>>>(nx1, nx2, gidx2, 512, 128, 64, 0.16f);
  }
  {
    long tot = (long)B * 128 * 64 * 160;  // kp=160 for cin=131
    k_group_sa<<<gs(tot, 256), 256, 0, stream>>>(nx1, nx2, gidx2, l1pts, Xbuf,
                                                 512, 128, 64, 128, 160, B);
  }
  long M2 = (long)B * 128 * 64;
  run_mlp(M2, 10, 3);                      // sa2 -> Ybuf [M2,256]
  k_maxpool<<<gs((long)B * 128 * 256, 256), 256, 0, stream>>>(Ybuf, l2pts,
                                                              (long)B * 128, 64, 256);

  // ---------------- SA3: group_all, 259->[256,512,1024] -------------------
  {
    long tot = (long)B * 128 * 288;       // kp=288 for cin=259
    k_group_all<<<gs(tot, 256), 256, 0, stream>>>(nx2, l2pts, Xbuf, 128, 256, 288, B);
  }
  long M3 = (long)B * 128;
  run_mlp(M3, 13, 3);                      // sa3 -> Ybuf [M3,1024]
  k_maxpool<<<gs((long)B * 1024, 256), 256, 0, stream>>>(Ybuf, l3pts, (long)B, 128, 1024);

  // ---------------- FP3: S=1 broadcast, 1280->[256,256] -------------------
  k_fp_build<<<gs((long)B * 128, 128), 128, 0, stream>>>(nx2, nx2, l2pts, l3pts,
                                                         Xbuf, B, 128, 1, 256, 1024, 1280);
  run_mlp((long)B * 128, 5, 2);            // -> Ybuf [B*128,256]
  k_copyf<<<gs((long)B * 128 * 256, 256), 256, 0, stream>>>(Ybuf, fp3out,
                                                            (long)B * 128 * 256);

  // ---------------- FP2: 3-NN 512<-128, 384->[256,128] --------------------
  k_fp_build<<<gs((long)B * 512, 128), 128, 0, stream>>>(nx1, nx2, l1pts, fp3out,
                                                         Xbuf, B, 512, 128, 128, 256, 384);
  run_mlp((long)B * 512, 3, 2);            // -> Ybuf [B*512,128]
  k_copyf<<<gs((long)B * 512 * 128, 256), 256, 0, stream>>>(Ybuf, fp2out,
                                                            (long)B * 512 * 128);

  // ---------------- FP1: 3-NN 4096<-512, 154->[128,128,128] ---------------
  k_p1_build<<<gs((long)B * N * 26, 256), 256, 0, stream>>>(cls_in, l0x, l0p, p1buf, B, N);
  k_fp_build<<<gs((long)B * N, 128), 128, 0, stream>>>(l0x, nx1, p1buf, fp2out,
                                                       Xbuf, B, N, 512, 26, 128, 160);
  run_mlp((long)B * N, 0, 3);              // -> Ybuf [B*N,128]

  // ---------------- output transpose (B,N,128)->(B,128,N) -----------------
  k_transpose_out<<<gs((long)B * 128 * N, 256), 256, 0, stream>>>(Ybuf, (float*)d_out, B, N);
}
